// DeformableConv2d_11003706212920
// MI455X (gfx1250) — matrix operand order repaired and checked by reading, not compiled, not run on hardware
//
#include <hip/hip_runtime.h>
#include <math.h>

// ---------------------------------------------------------------------------
// Modulated deformable conv v2 on MI455X (gfx1250), wave32 + WMMA f16.
//
//   K0a prep  : weights -> f16 "Bt" matrices [n][k], k = tap*128 + ci.
//   K0b stage : x (NCHW f32) -> xh (B,HW,C f16).  HWC layout makes each
//               8-wide K-run of an A fragment ONE aligned b128 load.
//   K1 omGEMM : offset/mask conv as M=32768,K=1152,N=32 WMMA GEMM.
//   K2 meta   : per (pixel,tap) clamped bilinear corners + 4 weights
//               premultiplied by 2*sigmoid(mod) and per-corner validity.
//   K3 GEMM   : deformable conv as M=32768,K=1152,N=128 WMMA GEMM; A built
//               from 4 corner b128 loads blended with v_pk_fma_f16.
//
// Roofline: ~12 GFLOP vs ~35 MB HBM (x L2-resident).  f16 WMMA (16x16x32,
// f32 accum) keeps both GEMMs near the memory roof; all padding/OOB handled
// arithmetically so EXEC is all-ones at every WMMA.
// ---------------------------------------------------------------------------

typedef __attribute__((ext_vector_type(16))) _Float16 v16h;
typedef __attribute__((ext_vector_type(8)))  _Float16 v8h;
typedef __attribute__((ext_vector_type(8)))  float    v8f;

#define BATCH 8
#define CIN   128
#define HH    64
#define WW    64
#define COUT  128
#define KK9   9
#define KTOT  1152          // CIN * KK9
#define NPIX  32768         // BATCH * HH * WW
#define HWSZ  4096          // HH * WW

__device__ __forceinline__ v8h splat8(_Float16 s) {
    v8h r = {s, s, s, s, s, s, s, s};
    return r;
}

__device__ __forceinline__ v16h cat16(v8h lo, v8h hi) {
    return __builtin_shufflevector(lo, hi, 0, 1, 2, 3, 4, 5, 6, 7,
                                   8, 9, 10, 11, 12, 13, 14, 15);
}

// ---------------------------------------------------------------------------
// K0a: weight prep.
//   btMain[co*1152 + tap*128 + ci] = (f16) w_reg[co][ci][tap]
//   btOM  [n *1152 + tap*128 + ci] = (f16) {w_off|w_mod|0}[n][ci][tap], n<32
// ---------------------------------------------------------------------------
__global__ void __launch_bounds__(256)
prep_weights_kernel(const float* __restrict__ w_reg,
                    const float* __restrict__ w_off,
                    const float* __restrict__ w_mod,
                    _Float16* __restrict__ btMain,
                    _Float16* __restrict__ btOM) {
    const int NMAIN = COUT * KTOT;        // 147456
    const int NOM   = 32 * KTOT;          // 36864
    int i = blockIdx.x * blockDim.x + threadIdx.x;
    if (i < NMAIN) {
        int co = i / KTOT;
        int k  = i - co * KTOT;
        int t  = k >> 7;                  // tap
        int ci = k & 127;
        btMain[i] = (_Float16)w_reg[(co * CIN + ci) * KK9 + t];
    } else if (i < NMAIN + NOM) {
        int j  = i - NMAIN;
        int n  = j / KTOT;
        int k  = j - n * KTOT;
        int t  = k >> 7;
        int ci = k & 127;
        float v = 0.0f;
        if (n < 18)      v = w_off[(n * CIN + ci) * KK9 + t];
        else if (n < 27) v = w_mod[((n - 18) * CIN + ci) * KK9 + t];
        btOM[n * KTOT + k] = (_Float16)v;
    }
}

// ---------------------------------------------------------------------------
// K0b: stage x NCHW f32 -> xh [b][hw][ci] f16 (channel-contiguous).
// ---------------------------------------------------------------------------
__global__ void __launch_bounds__(256)
stage_x_kernel(const float* __restrict__ x, _Float16* __restrict__ xh) {
    int i  = blockIdx.x * blockDim.x + threadIdx.x;   // 4194304 threads
    int sp = i & (HWSZ - 1);
    int ci = (i >> 12) & (CIN - 1);
    int b  = i >> 19;
    xh[(b * HWSZ + sp) * CIN + ci] = (_Float16)x[i];  // i == ((b*CIN+ci)*HWSZ+sp)
}

__device__ __forceinline__ float om_bias(const float* b_off, const float* b_mod, int n) {
    if (n < 18) return b_off[n];
    if (n < 27) return b_mod[n - 18];
    return 0.0f;
}

// ---------------------------------------------------------------------------
// K1: offset/mask conv as WMMA GEMM.  One wave = 16 pixels x 32 outputs.
//   omtmp[pix*32 + n] = conv_out[n] + bias[n]   (n in 0..26 meaningful)
// ---------------------------------------------------------------------------
__global__ void __launch_bounds__(256)
om_gemm_kernel(const _Float16* __restrict__ xh,
               const _Float16* __restrict__ btOM,
               const float* __restrict__ b_off,
               const float* __restrict__ b_mod,
               float* __restrict__ omtmp) {
    int gtid = blockIdx.x * blockDim.x + threadIdx.x;
    int wave = gtid >> 5;
    int lane = threadIdx.x & 31;
    int m    = lane & 15;
    int pix  = wave * 16 + m;                 // this lane's A-row pixel
    int b    = pix >> 12;
    int h    = (pix >> 6) & 63;
    int w    = pix & 63;
    const _Float16* xb2 = xh + b * (HWSZ * CIN);

    const int kbase = (lane >= 16) ? 8  : 0;  // A-layout K split (ISA 7.12.2)
    const int bbase = (lane >= 16) ? 16 : 0;  // unused; B loads also use kbase

    v8f acc0 = {};
    v8f acc1 = {};

    for (int t = 0; t < 9; ++t) {
        int yy = h - 1 + t / 3;
        int xx = w - 1 + t % 3;
        bool ok = (yy >= 0) && (yy < HH) && (xx >= 0) && (xx < WW);
        int sp = ok ? (yy * WW + xx) : 0;               // clamped address
        v8h oks = splat8(ok ? (_Float16)1.0f : (_Float16)0.0f);
        const _Float16* ps = xb2 + sp * CIN;
        for (int cg = 0; cg < 4; ++cg) {
            int cb = cg * 32 + kbase;
            v8h qlo = *(const v8h*)(ps + cb) * oks;      // unconditional loads
            v8h qhi = *(const v8h*)(ps + cb + 16) * oks;
            v16h a  = cat16(qlo, qhi);
            int kk = t * 128 + cg * 32;
            v16h bf0 = cat16(*(const v8h*)(btOM + (0 * 16 + m) * KTOT + kk + kbase),
                             *(const v8h*)(btOM + (0 * 16 + m) * KTOT + kk + kbase + 16));
            v16h bf1 = cat16(*(const v8h*)(btOM + (1 * 16 + m) * KTOT + kk + kbase),
                             *(const v8h*)(btOM + (1 * 16 + m) * KTOT + kk + kbase + 16));
            acc0 = __builtin_amdgcn_wmma_f32_16x16x32_f16(false, a, false, bf0,
                                                          (short)0, acc0, false, false);
            acc1 = __builtin_amdgcn_wmma_f32_16x16x32_f16(false, a, false, bf1,
                                                          (short)0, acc1, false, false);
        }
    }

    // Store with bias.  C layout: reg r, lane -> M = r + 8*(lane>=16), N = lane&15.
    float bias0 = om_bias(b_off, b_mod, m);
    float bias1 = om_bias(b_off, b_mod, 16 + m);
    int mhi = (lane >= 16) ? 8 : 0;
#pragma unroll
    for (int r = 0; r < 8; ++r) {
        int p2 = wave * 16 + r + mhi;
        omtmp[p2 * 32 + m]      = acc0[r] + bias0;
        omtmp[p2 * 32 + 16 + m] = acc1[r] + bias1;
    }
}

// ---------------------------------------------------------------------------
// K2: metadata.  One thread per (pixel, tap).
//   metaW: 4 bilinear weights * 2*sigmoid(mod) * per-corner validity
//   metaC: clamped (y0, y1, x0, x1)
// ---------------------------------------------------------------------------
__global__ void __launch_bounds__(256)
meta_kernel(const float* __restrict__ omtmp,
            float4* __restrict__ metaW,
            int4* __restrict__ metaC) {
    int i = blockIdx.x * blockDim.x + threadIdx.x;
    if (i >= NPIX * KK9) return;
    int pix = i / KK9;
    int t   = i - pix * KK9;
    int h   = (pix >> 6) & 63;
    int w   = pix & 63;
    const float* o = omtmp + pix * 32;
    float dy = o[2 * t];
    float dx = o[2 * t + 1];
    float mv = 2.0f / (1.0f + expf(-o[18 + t]));   // 2*sigmoid
    float py = (float)(h - 1 + t / 3) + dy;
    float px = (float)(w - 1 + t % 3) + dx;
    float y0f = floorf(py), x0f = floorf(px);
    float wy = py - y0f, wx = px - x0f;
    int y0 = (int)y0f, x0 = (int)x0f;
    int y1 = y0 + 1,   x1 = x0 + 1;
    float vy0 = (y0 >= 0 && y0 < HH) ? 1.0f : 0.0f;
    float vy1 = (y1 >= 0 && y1 < HH) ? 1.0f : 0.0f;
    float vx0 = (x0 >= 0 && x0 < WW) ? 1.0f : 0.0f;
    float vx1 = (x1 >= 0 && x1 < WW) ? 1.0f : 0.0f;
    float w00 = (1.0f - wy) * (1.0f - wx) * mv * vy0 * vx0;
    float w01 = (1.0f - wy) * wx          * mv * vy0 * vx1;
    float w10 = wy          * (1.0f - wx) * mv * vy1 * vx0;
    float w11 = wy          * wx          * mv * vy1 * vx1;
    metaW[i] = make_float4(w00, w01, w10, w11);
    metaC[i] = make_int4(min(max(y0, 0), HH - 1), min(max(y1, 0), HH - 1),
                         min(max(x0, 0), WW - 1), min(max(x1, 0), WW - 1));
}

// ---------------------------------------------------------------------------
// K3: main deformable-conv GEMM.  One wave = 16 pixels x 128 Cout.
//   A[m][k=tap*128+ci] = bilinear(xh[b,*,ci]) * modulator, blended in packed
//   f16 from 4 corner b128 loads.  36 K-chunks x 8 WMMAs.
// ---------------------------------------------------------------------------
__global__ void __launch_bounds__(256)
deform_gemm_kernel(const _Float16* __restrict__ xh,
                   const _Float16* __restrict__ btMain,
                   const float4* __restrict__ metaW,
                   const int4* __restrict__ metaC,
                   float* __restrict__ out) {
    int gtid = blockIdx.x * blockDim.x + threadIdx.x;
    int wave = gtid >> 5;
    int lane = threadIdx.x & 31;
    int m    = lane & 15;
    int pix  = wave * 16 + m;
    int b    = pix >> 12;
    const _Float16* xb2 = xh + b * (HWSZ * CIN);

    const int kbase = (lane >= 16) ? 8  : 0;
    const int bbase = (lane >= 16) ? 16 : 0;

    v8f acc[8] = {};

    for (int t = 0; t < 9; ++t) {
        float4 wg = metaW[pix * KK9 + t];
        int4   cc = metaC[pix * KK9 + t];
        v8h w00s = splat8((_Float16)wg.x);
        v8h w01s = splat8((_Float16)wg.y);
        v8h w10s = splat8((_Float16)wg.z);
        v8h w11s = splat8((_Float16)wg.w);
        const _Float16* p00 = xb2 + (cc.x * WW + cc.z) * CIN;
        const _Float16* p01 = xb2 + (cc.x * WW + cc.w) * CIN;
        const _Float16* p10 = xb2 + (cc.y * WW + cc.z) * CIN;
        const _Float16* p11 = xb2 + (cc.y * WW + cc.w) * CIN;
        for (int cg = 0; cg < 4; ++cg) {
            int cb = cg * 32 + kbase;
            // 8 aligned b128 loads: 4 corners x 2 K-runs of 8 channels
            v8h a00 = *(const v8h*)(p00 + cb);
            v8h a01 = *(const v8h*)(p01 + cb);
            v8h a10 = *(const v8h*)(p10 + cb);
            v8h a11 = *(const v8h*)(p11 + cb);
            v8h b00 = *(const v8h*)(p00 + cb + 16);
            v8h b01 = *(const v8h*)(p01 + cb + 16);
            v8h b10 = *(const v8h*)(p10 + cb + 16);
            v8h b11 = *(const v8h*)(p11 + cb + 16);
            // packed-f16 bilinear blend (v_pk_fma_f16)
            v8h qlo = a00 * w00s + a01 * w01s + a10 * w10s + a11 * w11s;
            v8h qhi = b00 * w00s + b01 * w01s + b10 * w10s + b11 * w11s;
            v16h a  = cat16(qlo, qhi);
            int kk = t * 128 + cg * 32;
#pragma unroll
            for (int n = 0; n < 8; ++n) {
                v16h bf = cat16(*(const v8h*)(btMain + (n * 16 + m) * KTOT + kk + kbase),
                                *(const v8h*)(btMain + (n * 16 + m) * KTOT + kk + kbase + 16));
                acc[n] = __builtin_amdgcn_wmma_f32_16x16x32_f16(false, a, false, bf,
                                                                (short)0, acc[n],
                                                                false, false);
            }
        }
    }

    // Store: out[b][co][h][w], C layout reg r -> M = r + 8*(lane>=16).
    int mhi = (lane >= 16) ? 8 : 0;
#pragma unroll
    for (int n = 0; n < 8; ++n) {
        int co = n * 16 + m;
#pragma unroll
        for (int r = 0; r < 8; ++r) {
            int p2  = wave * 16 + r + mhi;
            int bb  = p2 >> 12;
            int hw2 = p2 & 4095;
            out[(bb * COUT + co) * HWSZ + hw2] = acc[n][r];
        }
    }
}

// ---------------------------------------------------------------------------
// Host launch
// ---------------------------------------------------------------------------
extern "C" void kernel_launch(void* const* d_in, const int* in_sizes, int n_in,
                              void* d_out, int out_size, void* d_ws, size_t ws_size,
                              hipStream_t stream) {
    const float* x     = (const float*)d_in[0];
    const float* w_off = (const float*)d_in[1];
    const float* b_off = (const float*)d_in[2];
    const float* w_mod = (const float*)d_in[3];
    const float* b_mod = (const float*)d_in[4];
    const float* w_reg = (const float*)d_in[5];
    float* out = (float*)d_out;

    // Workspace carve-up (all offsets >=32B-aligned)
    char* ws = (char*)d_ws;
    const size_t SZ_XH     = (size_t)BATCH * HWSZ * CIN * sizeof(_Float16); // 8 MiB
    const size_t SZ_BTMAIN = (size_t)COUT * KTOT * sizeof(_Float16);        // 294912
    const size_t SZ_BTOM   = (size_t)32   * KTOT * sizeof(_Float16);        // 73728
    const size_t SZ_OMTMP  = (size_t)NPIX * 32 * sizeof(float);             // 4 MiB
    const size_t SZ_METAW  = (size_t)NPIX * KK9 * sizeof(float4);           // 4.5 MiB
    _Float16* xhp    = (_Float16*)(ws);
    _Float16* btMain = (_Float16*)(ws + SZ_XH);
    _Float16* btOM   = (_Float16*)(ws + SZ_XH + SZ_BTMAIN);
    float*    omtmp  = (float*)   (ws + SZ_XH + SZ_BTMAIN + SZ_BTOM);
    float4*   metaW  = (float4*)  (ws + SZ_XH + SZ_BTMAIN + SZ_BTOM + SZ_OMTMP);
    int4*     metaC  = (int4*)    (ws + SZ_XH + SZ_BTMAIN + SZ_BTOM + SZ_OMTMP + SZ_METAW);

    // K0a: weight prep (147456 + 36864 = 184320 threads)
    prep_weights_kernel<<<184320 / 256, 256, 0, stream>>>(w_reg, w_off, w_mod,
                                                          btMain, btOM);
    // K0b: stage x -> f16 HWC (4194304 threads)
    stage_x_kernel<<<(BATCH * CIN * HWSZ) / 256, 256, 0, stream>>>(x, xhp);
    // K1: offset/mask GEMM (2048 waves -> 256 blocks of 8 waves)
    om_gemm_kernel<<<256, 256, 0, stream>>>(xhp, btOM, b_off, b_mod, omtmp);
    // K2: metadata (32768*9 = 294912 threads)
    meta_kernel<<<294912 / 256, 256, 0, stream>>>(omtmp, metaW, metaC);
    // K3: main deformable GEMM (2048 waves -> 256 blocks)
    deform_gemm_kernel<<<256, 256, 0, stream>>>(xhp, btMain, metaW, metaC, out);
}
